// Model_29557964931141
// MI455X (gfx1250) — matrix operand order repaired and checked by reading, not compiled, not run on hardware
//
#include <hip/hip_runtime.h>
#include <hip/hip_bf16.h>
#include <math.h>

typedef __attribute__((ext_vector_type(16))) _Float16 v16h;
typedef __attribute__((ext_vector_type(8)))  float    v8f;

#define NN   8192   // nodes
#define FF   64     // node feat
#define EE   4096   // edge rows
#define FE   64     // edge feat
#define KK   512    // YT rows
#define O1   256
#define O2   64

#define WMMA_F16(A, B, C) \
    __builtin_amdgcn_wmma_f32_16x16x32_f16(false, (A), false, (B), (short)0, (C), false, false)

// ---------------- workspace layout (bytes) ----------------
// deg      f32[8192]      @ 0
// dvec     f32[8192]      @ 32768
// H2       f32[8192]      @ 65536
// svec     f32[8192]      @ 98304
// colsumET f32[64]        @ 131072
// rowsumW2 f32[64]        @ 131328
// w3y      f32[64]        @ 131584
// w4y      f32[64]        @ 131840
// scal     f32[2] WX,eyT  @ 132096
// vvec     f32[256]       @ 132352
// W1t      f16[256*64]    @ 135168
// Zt       f16[64*8192]   @ 262144
// AX       f32[8192*64]   @ 1572864   (ends 3670016)

__global__ __launch_bounds__(256) void k_zero(float* deg) {
    deg[blockIdx.x * 256 + threadIdx.x] = 0.0f;
}

// column sums of AT: each block handles 64 rows; thread t owns cols {4t..4t+3} + 1024*k
__global__ __launch_bounds__(256) void k_colsum(const float* __restrict__ AT,
                                                float* __restrict__ deg) {
    const int t = threadIdx.x;
    float acc[32];
#pragma unroll
    for (int k = 0; k < 32; ++k) acc[k] = 0.0f;
    const int r0 = blockIdx.x * 64;
    for (int r = 0; r < 64; ++r) {
        const float* row = AT + (size_t)(r0 + r) * NN + 4 * t;
#pragma unroll
        for (int k = 0; k < 8; ++k) {
            float4 v = *(const float4*)(row + k * 1024);
            acc[k * 4 + 0] += v.x;
            acc[k * 4 + 1] += v.y;
            acc[k * 4 + 2] += v.z;
            acc[k * 4 + 3] += v.w;
        }
    }
#pragma unroll
    for (int k = 0; k < 8; ++k) {
#pragma unroll
        for (int c = 0; c < 4; ++c)
            atomicAdd(&deg[4 * t + k * 1024 + c], acc[k * 4 + c]);
    }
}

// d = rsqrt(deg); Zt[f][j] = f16(d[j] * XT[j][f])  (transposed, f16)
__global__ __launch_bounds__(256) void k_prep(const float* __restrict__ deg,
                                              const float* __restrict__ XT,
                                              float* __restrict__ dvec,
                                              _Float16* __restrict__ Zt) {
    const int j = blockIdx.x * 256 + threadIdx.x;
    const float dj = rsqrtf(deg[j]);
    dvec[j] = dj;
    const float* x = XT + (size_t)j * FF;
#pragma unroll
    for (int f = 0; f < FF; ++f)
        Zt[(size_t)f * NN + j] = (_Float16)(dj * x[f]);
}

// W1t[n][k] = f16(W1[k][n])
__global__ __launch_bounds__(256) void k_w1t(const float* __restrict__ W1,
                                             _Float16* __restrict__ W1t) {
    const int k = blockIdx.x;      // 0..63
    const int n = threadIdx.x;     // 0..255
    W1t[(size_t)n * FF + k] = (_Float16)W1[(size_t)k * O1 + n];
}

// small reductions: colsumET, w3y=W3@YT, w4y=W4@YT, rowsumW2, WX, eyT
__global__ __launch_bounds__(256) void k_small(const float* __restrict__ ET,
                                               const float* __restrict__ W2,
                                               const float* __restrict__ W3,
                                               const float* __restrict__ W4,
                                               const float* __restrict__ YT,
                                               float* __restrict__ colsumET,
                                               float* __restrict__ rowsumW2,
                                               float* __restrict__ w3y,
                                               float* __restrict__ w4y,
                                               float* __restrict__ scal) {
    __shared__ float tmp[256];
    __shared__ float cs_s[64];
    __shared__ float rw_s[64];
    __shared__ float w4_s[64];
    const int t = threadIdx.x;
    const int f = t & 63, g = t >> 6;
    float acc = 0.0f;
    for (int e = g; e < EE; e += 4) acc += ET[(size_t)e * FE + f];
    tmp[t] = acc;
    __syncthreads();
    if (t < 64) {
        float c = tmp[t] + tmp[t + 64] + tmp[t + 128] + tmp[t + 192];
        cs_s[t] = c;
        colsumET[t] = c;
        float a3 = 0.0f, a4 = 0.0f;
        for (int k = 0; k < KK; ++k) {
            float y = YT[k];
            a3 += W3[(size_t)t * KK + k] * y;
            a4 += W4[(size_t)t * KK + k] * y;
        }
        w3y[t] = a3; w4y[t] = a4; w4_s[t] = a4;
        float rw = 0.0f;
        for (int o = 0; o < O1; ++o) rw += W2[(size_t)t * O1 + o];
        rowsumW2[t] = rw; rw_s[t] = rw;
    }
    __syncthreads();
    if (t == 0) {
        float wx = 0.0f, ey = 0.0f;
        for (int i = 0; i < 64; ++i) { wx += cs_s[i] * rw_s[i]; ey += cs_s[i] * w4_s[i]; }
        scal[0] = wx;   // WX
        scal[1] = ey;   // eyT
    }
}

// H2[m] = relu(dot(XT[m,:], w3y) + eyT + b2[m])
__global__ __launch_bounds__(256) void k_h2(const float* __restrict__ XT,
                                            const float* __restrict__ w3y,
                                            const float* __restrict__ scal,
                                            const float* __restrict__ b2,
                                            float* __restrict__ H2) {
    __shared__ float w[64];
    const int t = threadIdx.x;
    if (t < 64) w[t] = w3y[t];
    __syncthreads();
    const int m = blockIdx.x * 256 + t;
    const float* x = XT + (size_t)m * FF;
    float acc = 0.0f;
#pragma unroll
    for (int f = 0; f < FF; ++f) acc += x[f] * w[f];
    float h = acc + scal[1] + b2[m];
    H2[m] = h > 0.0f ? h : 0.0f;
}

// v[o] = dot(W5[o,:], H2)
__global__ __launch_bounds__(256) void k_v(const float* __restrict__ W5,
                                           const float* __restrict__ H2,
                                           float* __restrict__ vvec) {
    __shared__ float sh[256];
    const int o = blockIdx.x;
    const float* row = W5 + (size_t)o * NN;
    float acc = 0.0f;
    for (int m = threadIdx.x; m < NN; m += 256) acc += row[m] * H2[m];
    sh[threadIdx.x] = acc;
    __syncthreads();
    for (int s = 128; s > 0; s >>= 1) {
        if (threadIdx.x < s) sh[threadIdx.x] += sh[threadIdx.x + s];
        __syncthreads();
    }
    if (threadIdx.x == 0) vvec[o] = sh[0];
}

// ---- main GEMM: AX[m,:] = d[m] * (AT[m,:] @ Zt^T), one 16-row tile per wave ----
// K = 64 per iteration: 2 A fragments + 8 distinct B fragments. A liveness
// fence (empty asm) forces all fragments into distinct registers so all 16
// b128 loads are in flight before the 8 WMMAs drain behind partial waits.
__global__ __launch_bounds__(32) void k_main(const float* __restrict__ AT,
                                             const _Float16* __restrict__ Zt,
                                             const float* __restrict__ dvec,
                                             float* __restrict__ AX) {
    const int m0   = blockIdx.x * 16;
    const int lane = threadIdx.x;
    const int lo   = lane & 15;
    const int hi   = lane >> 4;

    const v8f vz = {0.f,0.f,0.f,0.f,0.f,0.f,0.f,0.f};
    v8f acc[4] = {vz, vz, vz, vz};

    const float*    arow = AT + (size_t)(m0 + lo) * NN;
    const _Float16* bp0  = Zt + (size_t)(0  + lo) * NN + hi * 16;
    const _Float16* bp1  = Zt + (size_t)(16 + lo) * NN + hi * 16;
    const _Float16* bp2  = Zt + (size_t)(32 + lo) * NN + hi * 16;
    const _Float16* bp3  = Zt + (size_t)(48 + lo) * NN + hi * 16;

    const int o1 = hi ? 16 : 0;
    const int o2 = hi ? 24 : 8;

    for (int k0 = 0; k0 < NN; k0 += 64) {
        // unconditional prefetch of the streaming A row ~4KB ahead
        // (masked address stays inside the row -> no branch in the hot loop)
        __builtin_prefetch(arow + ((k0 + 1024) & (NN - 1)), 0, 1);

        // A fragments (two 16x32 f16 tiles): lane<16 K={0..15}, lane>=16 K={16..31}
        float4 r0 = *(const float4*)(arow + k0 + o1);
        float4 r1 = *(const float4*)(arow + k0 + o1 + 4);
        float4 r2 = *(const float4*)(arow + k0 + o2);
        float4 r3 = *(const float4*)(arow + k0 + o2 + 4);
        float4 r4 = *(const float4*)(arow + k0 + 32 + o1);
        float4 r5 = *(const float4*)(arow + k0 + 32 + o1 + 4);
        float4 r6 = *(const float4*)(arow + k0 + 32 + o2);
        float4 r7 = *(const float4*)(arow + k0 + 32 + o2 + 4);

        // B fragments (32x16 f16): lane holds 16 contiguous K of its column
        v16h b00 = *(const v16h*)(bp0 + k0);
        v16h b01 = *(const v16h*)(bp1 + k0);
        v16h b02 = *(const v16h*)(bp2 + k0);
        v16h b03 = *(const v16h*)(bp3 + k0);
        v16h b10 = *(const v16h*)(bp0 + k0 + 32);
        v16h b11 = *(const v16h*)(bp1 + k0 + 32);
        v16h b12 = *(const v16h*)(bp2 + k0 + 32);
        v16h b13 = *(const v16h*)(bp3 + k0 + 32);

        v16h a0, a1;
        a0[0]=(_Float16)r0.x;  a0[1]=(_Float16)r0.y;  a0[2]=(_Float16)r0.z;  a0[3]=(_Float16)r0.w;
        a0[4]=(_Float16)r1.x;  a0[5]=(_Float16)r1.y;  a0[6]=(_Float16)r1.z;  a0[7]=(_Float16)r1.w;
        a0[8]=(_Float16)r2.x;  a0[9]=(_Float16)r2.y;  a0[10]=(_Float16)r2.z; a0[11]=(_Float16)r2.w;
        a0[12]=(_Float16)r3.x; a0[13]=(_Float16)r3.y; a0[14]=(_Float16)r3.z; a0[15]=(_Float16)r3.w;
        a1[0]=(_Float16)r4.x;  a1[1]=(_Float16)r4.y;  a1[2]=(_Float16)r4.z;  a1[3]=(_Float16)r4.w;
        a1[4]=(_Float16)r5.x;  a1[5]=(_Float16)r5.y;  a1[6]=(_Float16)r5.z;  a1[7]=(_Float16)r5.w;
        a1[8]=(_Float16)r6.x;  a1[9]=(_Float16)r6.y;  a1[10]=(_Float16)r6.z; a1[11]=(_Float16)r6.w;
        a1[12]=(_Float16)r7.x; a1[13]=(_Float16)r7.y; a1[14]=(_Float16)r7.z; a1[15]=(_Float16)r7.w;

        // Liveness fence: force all A/B fragments simultaneously live in
        // distinct register tuples so the allocator cannot serialize the
        // B loads through one shared tuple.
        asm volatile("" : "+v"(a0), "+v"(a1),
                          "+v"(b00), "+v"(b01), "+v"(b02), "+v"(b03),
                          "+v"(b10), "+v"(b11), "+v"(b12), "+v"(b13));

        acc[0] = WMMA_F16(a0, b00, acc[0]);
        acc[1] = WMMA_F16(a0, b01, acc[1]);
        acc[2] = WMMA_F16(a0, b02, acc[2]);
        acc[3] = WMMA_F16(a0, b03, acc[3]);
        acc[0] = WMMA_F16(a1, b10, acc[0]);
        acc[1] = WMMA_F16(a1, b11, acc[1]);
        acc[2] = WMMA_F16(a1, b12, acc[2]);
        acc[3] = WMMA_F16(a1, b13, acc[3]);
    }

    float4 dA = *(const float4*)(dvec + m0 + hi * 8);
    float4 dB = *(const float4*)(dvec + m0 + hi * 8 + 4);
    float dm[8] = {dA.x, dA.y, dA.z, dA.w, dB.x, dB.y, dB.z, dB.w};
#pragma unroll
    for (int r = 0; r < 8; ++r) {
        float* o = AX + (size_t)(m0 + hi * 8 + r) * FF + lo;
        o[0]  = acc[0][r] * dm[r];
        o[16] = acc[1][r] * dm[r];
        o[32] = acc[2][r] * dm[r];
        o[48] = acc[3][r] * dm[r];
    }
}

// ---- s[m] = relu(AX@W1 + WX + b1) . v   (H1 tile kept in registers) ----
__global__ __launch_bounds__(32) void k_s(const float* __restrict__ AX,
                                          const _Float16* __restrict__ W1t,
                                          const float* __restrict__ vvec,
                                          const float* __restrict__ scal,
                                          const float* __restrict__ b1,
                                          float* __restrict__ svec) {
    const int m0   = blockIdx.x * 16;
    const int lane = threadIdx.x;
    const int lo   = lane & 15;
    const int hi   = lane >> 4;

    const float* arow = AX + (size_t)(m0 + lo) * FF;
    v16h a[2];
#pragma unroll
    for (int t = 0; t < 2; ++t) {
        const int k0 = t * 32;
        const int o1 = k0 + (hi ? 16 : 0);
        const int o2 = k0 + (hi ? 24 : 8);
        float4 r0 = *(const float4*)(arow + o1);
        float4 r1 = *(const float4*)(arow + o1 + 4);
        float4 r2 = *(const float4*)(arow + o2);
        float4 r3 = *(const float4*)(arow + o2 + 4);
        v16h av;
        av[0]=(_Float16)r0.x;  av[1]=(_Float16)r0.y;  av[2]=(_Float16)r0.z;  av[3]=(_Float16)r0.w;
        av[4]=(_Float16)r1.x;  av[5]=(_Float16)r1.y;  av[6]=(_Float16)r1.z;  av[7]=(_Float16)r1.w;
        av[8]=(_Float16)r2.x;  av[9]=(_Float16)r2.y;  av[10]=(_Float16)r2.z; av[11]=(_Float16)r2.w;
        av[12]=(_Float16)r3.x; av[13]=(_Float16)r3.y; av[14]=(_Float16)r3.z; av[15]=(_Float16)r3.w;
        a[t] = av;
    }

    const float WX = scal[0];
    float4 bA = *(const float4*)(b1 + m0 + hi * 8);
    float4 bB = *(const float4*)(b1 + m0 + hi * 8 + 4);
    float bb[8] = {bA.x, bA.y, bA.z, bA.w, bB.x, bB.y, bB.z, bB.w};
    float rs[8] = {0.f, 0.f, 0.f, 0.f, 0.f, 0.f, 0.f, 0.f};

    const v8f vz = {0.f,0.f,0.f,0.f,0.f,0.f,0.f,0.f};
#pragma unroll
    for (int nc = 0; nc < 4; ++nc) {
        v8f acc[4] = {vz, vz, vz, vz};
#pragma unroll
        for (int t = 0; t < 2; ++t) {
#pragma unroll
            for (int j = 0; j < 4; ++j) {
                const int col = nc * 64 + j * 16 + lo;
                v16h b = *(const v16h*)(W1t + (size_t)col * FF + hi * 16 + t * 32);
                acc[j] = WMMA_F16(a[t], b, acc[j]);
            }
        }
#pragma unroll
        for (int j = 0; j < 4; ++j) {
            const float vn = vvec[nc * 64 + j * 16 + lo];
#pragma unroll
            for (int r = 0; r < 8; ++r) {
                float h = acc[j][r] + WX + bb[r];
                h = h > 0.0f ? h : 0.0f;
                rs[r] += h * vn;
            }
        }
    }
#pragma unroll
    for (int r = 0; r < 8; ++r) {
        rs[r] += __shfl_xor(rs[r], 1, 32);
        rs[r] += __shfl_xor(rs[r], 2, 32);
        rs[r] += __shfl_xor(rs[r], 4, 32);
        rs[r] += __shfl_xor(rs[r], 8, 32);
    }
    if (lo == 0) {
#pragma unroll
        for (int r = 0; r < 8; ++r) svec[m0 + hi * 8 + r] = rs[r];
    }
}

// out[m,o] = sigmoid(s[m] * W6[o])
__global__ __launch_bounds__(256) void k_final(const float* __restrict__ svec,
                                               const float* __restrict__ W6,
                                               float* __restrict__ out) {
    const int idx = blockIdx.x * 256 + threadIdx.x;
    const int m = idx >> 6;
    const int o = idx & 63;
    const float x = svec[m] * W6[o];
    out[idx] = 1.0f / (1.0f + expf(-x));
}

extern "C" void kernel_launch(void* const* d_in, const int* in_sizes, int n_in,
                              void* d_out, int out_size, void* d_ws, size_t ws_size,
                              hipStream_t stream) {
    const float* AT = (const float*)d_in[0];
    const float* XT = (const float*)d_in[1];
    const float* ET = (const float*)d_in[2];
    const float* YT = (const float*)d_in[3];
    const float* W1 = (const float*)d_in[4];
    const float* W2 = (const float*)d_in[5];
    const float* W3 = (const float*)d_in[6];
    const float* W4 = (const float*)d_in[7];
    const float* W5 = (const float*)d_in[8];
    const float* W6 = (const float*)d_in[9];
    const float* b1 = (const float*)d_in[10];
    const float* b2 = (const float*)d_in[11];
    float* out = (float*)d_out;

    char* ws = (char*)d_ws;
    float*     deg      = (float*)(ws + 0);
    float*     dvec     = (float*)(ws + 32768);
    float*     H2       = (float*)(ws + 65536);
    float*     svec     = (float*)(ws + 98304);
    float*     colsumET = (float*)(ws + 131072);
    float*     rowsumW2 = (float*)(ws + 131328);
    float*     w3y      = (float*)(ws + 131584);
    float*     w4y      = (float*)(ws + 131840);
    float*     scal     = (float*)(ws + 132096);
    float*     vvec     = (float*)(ws + 132352);
    _Float16*  W1t      = (_Float16*)(ws + 135168);
    _Float16*  Zt       = (_Float16*)(ws + 262144);
    float*     AX       = (float*)(ws + 1572864);

    k_zero  <<<NN / 256, 256, 0, stream>>>(deg);
    k_colsum<<<NN / 64, 256, 0, stream>>>(AT, deg);
    k_prep  <<<NN / 256, 256, 0, stream>>>(deg, XT, dvec, Zt);
    k_w1t   <<<FF, 256, 0, stream>>>(W1, W1t);
    k_small <<<1, 256, 0, stream>>>(ET, W2, W3, W4, YT, colsumET, rowsumW2, w3y, w4y, scal);
    k_h2    <<<NN / 256, 256, 0, stream>>>(XT, w3y, scal, b2, H2);
    k_v     <<<O1, 256, 0, stream>>>(W5, H2, vvec);
    k_main  <<<NN / 16, 32, 0, stream>>>(AT, Zt, dvec, AX);
    k_s     <<<NN / 16, 32, 0, stream>>>(AX, W1t, vvec, scal, b1, svec);
    k_final <<<(NN * O2) / 256, 256, 0, stream>>>(svec, W6, out);
}
